// Net_73229192397025
// MI455X (gfx1250) — compile-verified, matrix operand order checked by reading, not run on hardware
//
#include <hip/hip_runtime.h>
#include <hip/hip_bf16.h>
#include <stdint.h>

// ---------------------------------------------------------------------------
// MI455X (gfx1250) implementation. wave32, WMMA bf16 16x16x32 with f32 accum.
// GEMMs register-blocked 4 n-tiles/wave for 4x A-operand reuse.
// ---------------------------------------------------------------------------

typedef unsigned short ushort_t;
typedef __attribute__((ext_vector_type(16))) __bf16 v16bf;
typedef __attribute__((ext_vector_type(8)))  float  v8f;

#define EPS_ 1e-5f
constexpr int B_    = 8;
constexpr int NPTS_ = 2826;
constexpr int N_    = B_ * NPTS_;        // 22608 (multiple of 16)
constexpr int K_    = 10;
constexpr int DIL_  = 4;
constexpr int E_FULL = N_ * K_;          // 226080 (multiple of 16)
constexpr int E_DIL  = E_FULL / DIL_;    // 56520
constexpr int E_DILP = (E_DIL + 15) & ~15; // 56528

__device__ __forceinline__ ushort_t f2bf(float f) {
  uint32_t u = __float_as_uint(f);
  u += 0x7FFFu + ((u >> 16) & 1u);       // round-to-nearest-even
  return (ushort_t)(u >> 16);
}

union FragBF { uint4 q[2]; v16bf v; };

// ---------------------------------------------------------------------------
// 1) Brute-force kNN per graph (self-loop included; stable ties like top_k)
// ---------------------------------------------------------------------------
template <int D>
__global__ void knn_kernel(const float* __restrict__ X, int* __restrict__ nbr) {
  int gid = blockIdx.x * blockDim.x + threadIdx.x;
  if (gid >= N_) return;
  int base = (gid / NPTS_) * NPTS_;
  float q[D];
#pragma unroll
  for (int c = 0; c < D; ++c) q[c] = X[(size_t)gid * D + c];
  float bd[K_]; int bi[K_];
#pragma unroll
  for (int j = 0; j < K_; ++j) { bd[j] = 3.4e38f; bi[j] = base; }
  for (int m = 0; m < NPTS_; ++m) {
    const float* p = X + (size_t)(base + m) * D;
    float d2 = 0.f;
#pragma unroll
    for (int c = 0; c < D; ++c) { float t = q[c] - p[c]; d2 += t * t; }
    if (d2 < bd[K_ - 1]) {
      int j = K_ - 1;
      while (j > 0 && d2 < bd[j - 1]) { bd[j] = bd[j - 1]; bi[j] = bi[j - 1]; --j; }
      bd[j] = d2; bi[j] = base + m;
    }
  }
#pragma unroll
  for (int j = 0; j < K_; ++j) nbr[(size_t)gid * K_ + j] = bi[j];
}

// ---------------------------------------------------------------------------
// 2) Weight packing: fp32 [K,N] row-major -> bf16 per-lane WMMA B fragments.
//    Layout: [K/32][N/16][lane 0..31][i 0..15] with
//    k = kt*32 + (lane>>4)*8 + (i&7) + (i>>3)*16 ; n = tn*16 + (lane&15)
// ---------------------------------------------------------------------------
__global__ void pack_w_kernel(const float* __restrict__ W, ushort_t* __restrict__ Wp,
                              int Korig, int Norig, int Kpad, int Npad) {
  int t = blockIdx.x * blockDim.x + threadIdx.x;
  int ntn = Npad >> 4;
  int total = (Kpad >> 5) * ntn * 512;
  if (t >= total) return;
  int i    = t & 15;
  int lane = (t >> 4) & 31;
  int tile = t >> 9;
  int tn   = tile % ntn;
  int kt   = tile / ntn;
  int k = (kt << 5) + ((lane >> 4) << 3) + (i & 7) + ((i >> 3) << 4);
  int n = (tn << 4) + (lane & 15);
  float v = (k < Korig && n < Norig) ? W[(size_t)k * Norig + n] : 0.f;
  Wp[t] = f2bf(v);
}

// ---------------------------------------------------------------------------
// 3) WMMA GEMM: out = BN(ReLU(A@W + b)) (or plain A@W+b).
//    NT n-tiles per wave: A fragment loaded once per k-step, NT WMMAs issued.
//    A: bf16 row-major [M, Kd]; Wp: packed as above; outF fp32 / outB bf16.
// ---------------------------------------------------------------------------
template <int NT>
__global__ void wmma_gemm_kernel(const ushort_t* __restrict__ A,
                                 const ushort_t* __restrict__ Wp,
                                 const float* __restrict__ bias,
                                 const float* __restrict__ g,
                                 const float* __restrict__ bt,
                                 const float* __restrict__ mu,
                                 const float* __restrict__ var,
                                 float* __restrict__ outF,
                                 ushort_t* __restrict__ outB,
                                 int M, int Kd, int Nd, int Nvalid, int relu_bn) {
  int lane = threadIdx.x & 31;
  int wave = threadIdx.x >> 5;
  int ntn  = Nd >> 4;          // total 16-wide n tiles
  int ng   = ntn / NT;         // n-tile groups per output row
  long tile  = (long)blockIdx.x * (blockDim.x >> 5) + wave;
  long tiles = (long)(M >> 4) * ng;
  if (tile >= tiles) return;
  int  tn0 = (int)(tile % ng) * NT;
  long tm  = tile / ng;

  int m    = lane & 15;
  int half = lane >> 4;
  const ushort_t* arow  = A + ((size_t)(tm * 16 + m)) * Kd + half * 8;
  const ushort_t* bbase = Wp + (size_t)tn0 * 512 + (size_t)lane * 16;
  size_t bstride = (size_t)ntn * 512;

  v8f acc[NT] = {};
  for (int k0 = 0; k0 < Kd; k0 += 32) {
    FragBF a;
    a.q[0] = *(const uint4*)(arow + k0);
    a.q[1] = *(const uint4*)(arow + k0 + 16);
    if (k0 + 32 < Kd) __builtin_prefetch(arow + k0 + 32, 0, 1);
    const ushort_t* bp = bbase + (size_t)(k0 >> 5) * bstride;
#pragma unroll
    for (int nt = 0; nt < NT; ++nt) {
      FragBF b;
      b.q[0] = *(const uint4*)(bp + (size_t)nt * 512);
      b.q[1] = *(const uint4*)(bp + (size_t)nt * 512 + 8);
      acc[nt] = __builtin_amdgcn_wmma_f32_16x16x32_bf16(
          false, a.v, false, b.v, (short)0, acc[nt], false, false);
    }
  }

#pragma unroll
  for (int nt = 0; nt < NT; ++nt) {
    int chan = (tn0 + nt) * 16 + m;            // N index is fixed per lane
    if (chan >= Nvalid) continue;
    float bs = bias[chan];
    float s = 1.f, t = 0.f;
    if (relu_bn) {
      float rs = rsqrtf(var[chan] + EPS_);
      s = g[chan] * rs;
      t = bt[chan] - mu[chan] * s;
    }
#pragma unroll
    for (int r = 0; r < 8; ++r) {
      long row = tm * 16 + r + half * 8;       // C layout: VGPR r -> M=r / r+8
      float h = acc[nt][r] + bs;
      if (relu_bn) { h = fmaxf(h, 0.f); h = h * s + t; }
      size_t off = (size_t)row * Nd + chan;
      if (outF) outF[off] = h;
      if (outB) outB[off] = f2bf(h);
    }
  }
}

// ---------------------------------------------------------------------------
// 4) Edge feature builder: row r -> [x_i | x_j - x_i | 0-pad] in bf16
// ---------------------------------------------------------------------------
__global__ void edge_feat_kernel(const float* __restrict__ F, const int* __restrict__ nbr,
                                 ushort_t* __restrict__ A, int d, int Kpad, int dil, int rows) {
  int r = blockIdx.x * blockDim.x + threadIdx.x;
  if (r >= rows) return;
  ushort_t* out = A + (size_t)r * Kpad;
  long e = (long)r * dil;
  if (e >= (long)E_FULL) {                     // padded rows -> zeros
    for (int c = 0; c < Kpad; ++c) out[c] = 0;
    return;
  }
  int i = (int)(e / K_);
  int j = (int)(e % K_);
  int jn = nbr[(size_t)i * K_ + j];
  const float* xi = F + (size_t)i * d;
  const float* xj = F + (size_t)jn * d;
  for (int c = 0; c < d; ++c) out[c] = f2bf(xi[c]);
  for (int c = 0; c < d; ++c) out[d + c] = f2bf(xj[c] - xi[c]);
  for (int c = 2 * d; c < Kpad; ++c) out[c] = 0;
}

// ---------------------------------------------------------------------------
// 5) Per-node segment max (edges for dst i are rows ceil(iK/dil)..(iK+K-1)/dil)
// ---------------------------------------------------------------------------
__global__ void segmax_kernel(const float* __restrict__ msg, float* __restrict__ feat,
                              int colOff, int dil) {
  int t = blockIdx.x * blockDim.x + threadIdx.x;
  if (t >= N_ * 64) return;
  int i = t >> 6, c = t & 63;
  long e0 = (long)i * K_;
  long r0 = (e0 + dil - 1) / dil;
  long r1 = (e0 + K_ - 1) / dil;
  float v = -3.4e38f;
  for (long r = r0; r <= r1; ++r) v = fmaxf(v, msg[(size_t)r * 64 + c]);
  feat[(size_t)i * 256 + colOff + c] = v;
}

__global__ void f2b_kernel(const float* __restrict__ in, ushort_t* __restrict__ out, size_t n) {
  size_t i = (size_t)blockIdx.x * blockDim.x + threadIdx.x;
  if (i < n) out[i] = f2bf(in[i]);
}

// gf [N,1024]: max over channels for first 778 points of each graph
__global__ void hand_kernel(const float* __restrict__ gf, float* __restrict__ hand) {
  int t = blockIdx.x * blockDim.x + threadIdx.x;
  if (t >= B_ * 778) return;
  int b = t / 778, p = t % 778;
  const float* row = gf + (size_t)(b * NPTS_ + p) * 1024;
  float v = -3.4e38f;
  for (int c = 0; c < 1024; ++c) v = fmaxf(v, row[c]);
  hand[t] = v;
}

// gf [N,1024]: max over nodes per graph -> gmax [B,1024]
__global__ void gmax_kernel(const float* __restrict__ gf, float* __restrict__ gmax) {
  int t = blockIdx.x * blockDim.x + threadIdx.x;
  if (t >= B_ * 1024) return;
  int b = t >> 10, c = t & 1023;
  float v = -3.4e38f;
  for (int p = 0; p < NPTS_; ++p) v = fmaxf(v, gf[(size_t)(b * NPTS_ + p) * 1024 + c]);
  gmax[t] = v;
}

// cat bf16 [N,1312] = [gmax(1024) | onehot(28) | feat(256) | zeros(4)]
__global__ void cat_kernel(const float* __restrict__ gmax, const float* __restrict__ onehot,
                           const float* __restrict__ feat, ushort_t* __restrict__ cat) {
  size_t t = (size_t)blockIdx.x * blockDim.x + threadIdx.x;
  if (t >= (size_t)N_ * 1312) return;
  int n = (int)(t / 1312), c = (int)(t % 1312);
  int b = n / NPTS_;
  float v;
  if      (c < 1024) v = gmax[(size_t)b * 1024 + c];
  else if (c < 1052) v = onehot[(size_t)n * 28 + (c - 1024)];
  else if (c < 1308) v = feat[(size_t)n * 256 + (c - 1052)];
  else               v = 0.f;
  cat[t] = f2bf(v);
}

__global__ void lsm_kernel(const float* __restrict__ logits, float* __restrict__ out) {
  int n = blockIdx.x * blockDim.x + threadIdx.x;
  if (n >= N_) return;
  const float* row = logits + (size_t)n * 16;
  float m = -3.4e38f;
  for (int c = 0; c < 10; ++c) m = fmaxf(m, row[c]);
  float s = 0.f;
  for (int c = 0; c < 10; ++c) s += __expf(row[c] - m);
  float ls = __logf(s);
  for (int c = 0; c < 10; ++c) out[(size_t)n * 10 + c] = row[c] - m - ls;
}

// ---------------------------------------------------------------------------
// 6) mano branch: 8 rows, tiny -> one LDS kernel per graph
// ---------------------------------------------------------------------------
struct ManoParams {
  const float *W1,*b1,*g1,*t1,*m1,*v1;   // 778->256
  const float *W2,*b2,*g2,*t2,*m2,*v2;   // 256->128
  const float *W3,*b3,*g3,*t3,*m3,*v3;   // 128->64
  const float *W4,*b4;                   // 64->15
};

__global__ void mano_kernel(const float* __restrict__ hand, ManoParams P,
                            float* __restrict__ out) {
  __shared__ float sin_[784];
  __shared__ float sout_[256];
  int b = blockIdx.x, tid = threadIdx.x, nt = blockDim.x;
  for (int k = tid; k < 778; k += nt) sin_[k] = hand[b * 778 + k];
  __syncthreads();
#define MANO_LAYER(Wp, bp, gp, btp, mp, vp, CIN, COUT)                         \
  for (int c = tid; c < COUT; c += nt) {                                       \
    float a = bp[c];                                                           \
    for (int k = 0; k < CIN; ++k) a += sin_[k] * Wp[(size_t)k * COUT + c];     \
    a = fmaxf(a, 0.f);                                                         \
    a = (a - mp[c]) * rsqrtf(vp[c] + EPS_) * gp[c] + btp[c];                   \
    sout_[c] = a;                                                              \
  }                                                                            \
  __syncthreads();                                                             \
  for (int k = tid; k < COUT; k += nt) sin_[k] = sout_[k];                     \
  __syncthreads();
  MANO_LAYER(P.W1, P.b1, P.g1, P.t1, P.m1, P.v1, 778, 256)
  MANO_LAYER(P.W2, P.b2, P.g2, P.t2, P.m2, P.v2, 256, 128)
  MANO_LAYER(P.W3, P.b3, P.g3, P.t3, P.m3, P.v3, 128, 64)
#undef MANO_LAYER
  for (int c = tid; c < 15; c += nt) {
    float a = P.b4[c];
    for (int k = 0; k < 64; ++k) a += sin_[k] * P.W4[(size_t)k * 15 + c];
    out[(size_t)N_ * 10 + b * 15 + c] = a;
  }
}

// ---------------------------------------------------------------------------
// Host side
// ---------------------------------------------------------------------------
struct Layer { const float *W,*b,*g,*bt,*m,*v; };

static void run_pack(hipStream_t st, const float* W, ushort_t* Wp,
                     int Korig, int Norig, int Kpad, int Npad) {
  int total = (Kpad / 32) * (Npad / 16) * 512;
  pack_w_kernel<<<(total + 255) / 256, 256, 0, st>>>(W, Wp, Korig, Norig, Kpad, Npad);
}

static void run_gemm(hipStream_t st, const ushort_t* A, const ushort_t* Wp, const Layer& L,
                     float* outF, ushort_t* outB, int M, int Kpad, int Npad,
                     int Nvalid, int relu_bn) {
  int ntn = Npad / 16;
  if ((ntn & 3) == 0) {                        // 4 n-tiles per wave: 4x A reuse
    long tiles = (long)(M / 16) * (ntn / 4);
    int blocks = (int)((tiles + 7) / 8);       // 8 waves (256 threads) per block
    wmma_gemm_kernel<4><<<blocks, 256, 0, st>>>(A, Wp, L.b, L.g, L.bt, L.m, L.v,
                                                outF, outB, M, Kpad, Npad, Nvalid, relu_bn);
  } else {
    long tiles = (long)(M / 16) * ntn;
    int blocks = (int)((tiles + 7) / 8);
    wmma_gemm_kernel<1><<<blocks, 256, 0, st>>>(A, Wp, L.b, L.g, L.bt, L.m, L.v,
                                                outF, outB, M, Kpad, Npad, Nvalid, relu_bn);
  }
}

extern "C" void kernel_launch(void* const* d_in, const int* in_sizes, int n_in,
                              void* d_out, int out_size, void* d_ws, size_t ws_size,
                              hipStream_t stream) {
  (void)in_sizes; (void)n_in; (void)out_size; (void)ws_size;
  auto in_f = [&](int i) { return (const float*)d_in[i]; };

  // Leaf order: setup_inputs() dict insertion order, recursively.
  // Per MLP layer: W, b, g, bt, m, v.
  const float* x      = in_f(0);   // [N,25]
  const float* pos    = in_f(1);   // [N,3]
  const float* onehot = in_f(2);   // [N,28]
  auto mk = [&](int base) {
    Layer L; L.W = in_f(base); L.b = in_f(base+1); L.g = in_f(base+2);
    L.bt = in_f(base+3); L.m = in_f(base+4); L.v = in_f(base+5); return L;
  };
  Layer c1a = mk(4),  c1b = mk(10);
  Layer c2a = mk(16), c2b = mk(22);
  Layer c3a = mk(28), c3b = mk(34);
  Layer c4a = mk(40), c4b = mk(46);
  Layer l1a = mk(52), l1b = mk(58);
  Layer h1  = mk(64), h2  = mk(70);
  Layer hl; hl.W = in_f(76); hl.b = in_f(77); hl.g = hl.bt = hl.m = hl.v = nullptr;
  ManoParams MP;
  { Layer m1 = mk(78), m2 = mk(84), m3 = mk(90);
    MP.W1=m1.W; MP.b1=m1.b; MP.g1=m1.g; MP.t1=m1.bt; MP.m1=m1.m; MP.v1=m1.v;
    MP.W2=m2.W; MP.b2=m2.b; MP.g2=m2.g; MP.t2=m2.bt; MP.m2=m2.m; MP.v2=m2.v;
    MP.W3=m3.W; MP.b3=m3.b; MP.g3=m3.g; MP.t3=m3.bt; MP.m3=m3.m; MP.v3=m3.v;
    MP.W4=in_f(96); MP.b4=in_f(97); }

  // ---- workspace carve (needs ~226 MB) ----
  char* ws = (char*)d_ws;
  size_t off = 0;
  auto take = [&](size_t bytes) { size_t o = off; off = (off + bytes + 255) & ~(size_t)255; return o; };
  int* nbr_pos = (int*)(ws + take((size_t)N_ * K_ * 4));
  int* nbr_x   = (int*)(ws + take((size_t)N_ * K_ * 4));
  ushort_t* wp_c1a = (ushort_t*)(ws + take(32   * 64   * 2));
  ushort_t* wp_c1b = (ushort_t*)(ws + take(64   * 64   * 2));
  ushort_t* wp_c2a = (ushort_t*)(ws + take(32   * 64   * 2));
  ushort_t* wp_c2b = (ushort_t*)(ws + take(64   * 64   * 2));
  ushort_t* wp_c3a = (ushort_t*)(ws + take(64   * 64   * 2));
  ushort_t* wp_c3b = (ushort_t*)(ws + take(64   * 64   * 2));
  ushort_t* wp_c4a = (ushort_t*)(ws + take(64   * 64   * 2));
  ushort_t* wp_c4b = (ushort_t*)(ws + take(64   * 64   * 2));
  ushort_t* wp_l1a = (ushort_t*)(ws + take(256  * 256  * 2));
  ushort_t* wp_l1b = (ushort_t*)(ws + take(256  * 1024 * 2));
  ushort_t* wp_h1  = (ushort_t*)(ws + take(1312 * 256  * 2));
  ushort_t* wp_h2  = (ushort_t*)(ws + take(256  * 128  * 2));
  ushort_t* wp_hl  = (ushort_t*)(ws + take(128  * 16   * 2));
  float*    feat   = (float*)   (ws + take((size_t)N_ * 256 * 4));
  ushort_t* featB  = (ushort_t*)(ws + take((size_t)N_ * 256 * 2));
  ushort_t* t1buf  = (ushort_t*)(ws + take((size_t)N_ * 256 * 2));
  float*    hand   = (float*)   (ws + take((size_t)B_ * 778 * 4));
  float*    gmaxb  = (float*)   (ws + take((size_t)B_ * 1024 * 4));
  ushort_t* catB   = (ushort_t*)(ws + take((size_t)N_ * 1312 * 2));
  // Aliased region R1: stage A (edge pipeline) then stage B (gf + head)
  constexpr size_t A_BY  = (size_t)E_FULL * 64 * 2;   // 28,938,240
  constexpr size_t H_BY  = (size_t)E_FULL * 64 * 2;
  constexpr size_t MSG_BY= (size_t)E_FULL * 64 * 4;   // 57,876,480
  constexpr size_t GF_BY = (size_t)N_ * 1024 * 4;     // 92,602,368
  constexpr size_t O1_BY = (size_t)N_ * 256 * 2;
  constexpr size_t O2_BY = (size_t)N_ * 128 * 2;
  constexpr size_t LG_BY = (size_t)N_ * 16 * 4;
  size_t r1 = take((A_BY + H_BY + MSG_BY > GF_BY + O1_BY + O2_BY + LG_BY)
                       ? (A_BY + H_BY + MSG_BY) : (GF_BY + O1_BY + O2_BY + LG_BY));
  ushort_t* Aedge = (ushort_t*)(ws + r1);
  ushort_t* h1e   = (ushort_t*)(ws + r1 + A_BY);
  float*    msg   = (float*)   (ws + r1 + A_BY + H_BY);
  float*    gf    = (float*)   (ws + r1);
  ushort_t* o1    = (ushort_t*)(ws + r1 + GF_BY);
  ushort_t* o2    = (ushort_t*)(ws + r1 + GF_BY + O1_BY);
  float*    logit = (float*)   (ws + r1 + GF_BY + O1_BY + O2_BY);

  // ---- 1. kNN graphs ----
  knn_kernel<3> <<<(N_ + 255) / 256, 256, 0, stream>>>(pos, nbr_pos);
  knn_kernel<25><<<(N_ + 255) / 256, 256, 0, stream>>>(x,   nbr_x);

  // ---- 2. pack all GEMM weights to bf16 WMMA fragments ----
  run_pack(stream, c1a.W, wp_c1a, 6,   64,   32,   64);
  run_pack(stream, c1b.W, wp_c1b, 64,  64,   64,   64);
  run_pack(stream, c2a.W, wp_c2a, 6,   64,   32,   64);
  run_pack(stream, c2b.W, wp_c2b, 64,  64,   64,   64);
  run_pack(stream, c3a.W, wp_c3a, 50,  64,   64,   64);
  run_pack(stream, c3b.W, wp_c3b, 64,  64,   64,   64);
  run_pack(stream, c4a.W, wp_c4a, 50,  64,   64,   64);
  run_pack(stream, c4b.W, wp_c4b, 64,  64,   64,   64);
  run_pack(stream, l1a.W, wp_l1a, 256, 256,  256,  256);
  run_pack(stream, l1b.W, wp_l1b, 256, 1024, 256,  1024);
  run_pack(stream, h1.W,  wp_h1,  1308,256,  1312, 256);
  run_pack(stream, h2.W,  wp_h2,  256, 128,  256,  128);
  run_pack(stream, hl.W,  wp_hl,  128, 10,   128,  16);

  // ---- 3. four EdgeConvs (build -> GEMM -> GEMM -> segment-max) ----
  auto conv = [&](const float* F, int d, int Kpad, const int* nbr, int dil, int rows,
                  const Layer& La, const ushort_t* Wa, const Layer& Lb, const ushort_t* Wb,
                  int colOff) {
    edge_feat_kernel<<<(rows + 255) / 256, 256, 0, stream>>>(F, nbr, Aedge, d, Kpad, dil, rows);
    run_gemm(stream, Aedge, Wa, La, nullptr, h1e, rows, Kpad, 64, 64, 1);
    run_gemm(stream, h1e,   Wb, Lb, msg, nullptr, rows, 64,   64, 64, 1);
    segmax_kernel<<<(N_ * 64 + 255) / 256, 256, 0, stream>>>(msg, feat, colOff, dil);
  };
  conv(pos, 3,  32, nbr_pos, 1,    E_FULL, c1a, wp_c1a, c1b, wp_c1b, 0);
  conv(pos, 3,  32, nbr_pos, DIL_, E_DILP, c2a, wp_c2a, c2b, wp_c2b, 64);
  conv(x,   25, 64, nbr_x,   1,    E_FULL, c3a, wp_c3a, c3b, wp_c3b, 128);
  conv(x,   25, 64, nbr_x,   DIL_, E_DILP, c4a, wp_c4a, c4b, wp_c4b, 192);

  // ---- 4. lin1: 256 -> 256 -> 1024 (gf overwrites the edge scratch region) ----
  f2b_kernel<<<(int)(((size_t)N_ * 256 + 255) / 256), 256, 0, stream>>>(feat, featB, (size_t)N_ * 256);
  run_gemm(stream, featB, wp_l1a, l1a, nullptr, t1buf, N_, 256, 256,  256,  1);
  run_gemm(stream, t1buf, wp_l1b, l1b, gf, nullptr,   N_, 256, 1024, 1024, 1);

  // ---- 5. reductions + mano branch ----
  hand_kernel<<<(B_ * 778 + 255) / 256, 256, 0, stream>>>(gf, hand);
  mano_kernel<<<B_, 256, 0, stream>>>(hand, MP, (float*)d_out);
  gmax_kernel<<<(B_ * 1024 + 255) / 256, 256, 0, stream>>>(gf, gmaxb);

  // ---- 6. head: cat(1312) -> 256 -> 128 -> 10 -> log_softmax ----
  cat_kernel<<<(int)(((size_t)N_ * 1312 + 255) / 256), 256, 0, stream>>>(gmaxb, onehot, feat, catB);
  run_gemm(stream, catB, wp_h1, h1, nullptr, o1, N_, 1312, 256, 256, 1);
  run_gemm(stream, o1,   wp_h2, h2, nullptr, o2, N_, 256,  128, 128, 1);
  run_gemm(stream, o2,   wp_hl, hl, logit, nullptr, N_, 128, 16, 10, 0);
  lsm_kernel<<<(N_ + 255) / 256, 256, 0, stream>>>(logit, (float*)d_out);
}
